// RWKVBlock_7352984010810
// MI455X (gfx1250) — matrix operand order repaired and checked by reading, not compiled, not run on hardware
//
#include <hip/hip_runtime.h>
#include <hip/hip_bf16.h>
#include <math.h>

// ---------------------------------------------------------------------------
// RWKV5-style block for MI455X (gfx1250, wave32, WMMA).
// All eight GEMMs run on v_wmma_f32_16x16x32_bf16 with f32 accumulation and
// fused epilogues. GEMM dims are template constants so epilogue stores fold
// into immediate offsets. Tile staging is double-buffered; when the toolchain
// declares the gfx1250 async global->LDS builtins we use them (ASYNCcnt +
// s_wait_asynccnt), otherwise a regs->LDS double buffer.
// ---------------------------------------------------------------------------

typedef __attribute__((ext_vector_type(16))) __bf16 bf16x16;
typedef __attribute__((ext_vector_type(8)))  __bf16 bf16x8;
typedef __attribute__((ext_vector_type(8)))  float  f32x8;
typedef __attribute__((ext_vector_type(4)))  int    i32x4;

#if defined(__has_builtin)
#if __has_builtin(__builtin_amdgcn_global_load_async_to_lds_b128) && \
    __has_builtin(__builtin_amdgcn_s_wait_asynccnt)
#define GEMM_ASYNC 1
#endif
#endif
#ifndef GEMM_ASYNC
#define GEMM_ASYNC 0
#endif

namespace {
constexpr int kB = 4, kT = 2048, kC = 1024, kH = 16, kS = 64, kFFN = 4096;
constexpr int kNTok = kB * kT;               // 8192 tokens
constexpr float kEps = 1e-5f;

// ---------------- GEMM: C[N x M] = A[N x K] (bf16) @ Bt[M x K]^T (bf16) -----
// Block tile 128x128, BK=32, 256 threads = 8 waves; each wave owns 64x32.
constexpr int BM = 128, BN = 128, BK = 32;
constexpr int LDT = 40;  // padded LDS row stride (bf16 elems), keeps 16B align

// EPI: 0 plain f32 | 1 silu f32 | 2 sigmoid f32 | 3 relu^2 -> bf16
//      4 out = res0 + acc (f32) | 5 out = res0 + res1*acc (f32)
template <int EPI, int K, int M>
__global__ __launch_bounds__(256, 2) void gemm_bf16_kernel(
    const __bf16* __restrict__ A, const __bf16* __restrict__ Bt,
    const float* __restrict__ res0, const float* __restrict__ res1,
    void* __restrict__ OutP) {
  __shared__ __bf16 As[2][BM * LDT];
  __shared__ __bf16 Bs[2][BN * LDT];

  const int tid  = threadIdx.x;
  const int lane = tid & 31;
  const int wave = tid >> 5;
  const int row0 = blockIdx.y * BM;          // token rows
  const int col0 = blockIdx.x * BN;          // output features
  const int wrow = (wave >> 2) * 64;         // 0 / 64
  const int wcol = (wave & 3) * 32;          // 0..96

  f32x8 acc[4][2];
#pragma unroll
  for (int i = 0; i < 4; ++i)
#pragma unroll
    for (int j = 0; j < 2; ++j)
#pragma unroll
      for (int e = 0; e < 8; ++e) acc[i][j][e] = 0.f;

#if GEMM_ASYNC
  // ---- gfx1250 async global->LDS staging (ASYNCcnt) ----
  auto stage = [&](int k0, int buf) {
#pragma unroll
    for (int it = 0; it < 2; ++it) {
      const int idx = (it * 256 + tid) * 8;
      const int r = idx >> 5;
      const int c = idx & 31;
      __builtin_amdgcn_global_load_async_to_lds_b128(
          (__attribute__((address_space(1))) i32x4*)(A +
              (size_t)(row0 + r) * K + k0 + c),
          (__attribute__((address_space(3))) i32x4*)(&As[buf][r * LDT + c]), 0,
          0);
      __builtin_amdgcn_global_load_async_to_lds_b128(
          (__attribute__((address_space(1))) i32x4*)(Bt +
              (size_t)(col0 + r) * K + k0 + c),
          (__attribute__((address_space(3))) i32x4*)(&Bs[buf][r * LDT + c]), 0,
          0);
    }
  };
  stage(0, 0);
  __builtin_amdgcn_s_wait_asynccnt(0);
  __syncthreads();
#else
  // ---- fallback: regs double buffer ----
  bf16x8 ra[2], rb[2];
  auto load_tile = [&](int k0) {
#pragma unroll
    for (int it = 0; it < 2; ++it) {
      const int idx = (it * 256 + tid) * 8;
      const int r = idx >> 5;
      const int c = idx & 31;
      ra[it] = *(const bf16x8*)(A + (size_t)(row0 + r) * K + k0 + c);
      rb[it] = *(const bf16x8*)(Bt + (size_t)(col0 + r) * K + k0 + c);
    }
  };
  auto store_tile = [&](int buf) {
#pragma unroll
    for (int it = 0; it < 2; ++it) {
      const int idx = (it * 256 + tid) * 8;
      const int r = idx >> 5;
      const int c = idx & 31;
      *(bf16x8*)(&As[buf][r * LDT + c]) = ra[it];
      *(bf16x8*)(&Bs[buf][r * LDT + c]) = rb[it];
    }
  };
  load_tile(0);
  store_tile(0);
  __syncthreads();
#endif

  int cur = 0;
  for (int k0 = 0; k0 < K; k0 += BK) {
    const bool more = (k0 + BK) < K;
#if GEMM_ASYNC
    if (more) stage(k0 + BK, cur ^ 1);
#else
    if (more) load_tile(k0 + BK);
#endif
    // A fragments: ISA 16-bit 16x32 layout. Lane half selects K base.
    const int arow = wrow + (lane & 15);
    const int akb  = (lane < 16) ? 0 : 8;    // K {0..7,16..23} / {8..15,24..31}
    bf16x16 afrag[4];
#pragma unroll
    for (int i = 0; i < 4; ++i) {
      const __bf16* p = &As[cur][(arow + i * 16) * LDT + akb];
      bf16x8 lo = *(const bf16x8*)(p);
      bf16x8 hi = *(const bf16x8*)(p + 16);
      afrag[i] = __builtin_shufflevector(lo, hi, 0, 1, 2, 3, 4, 5, 6, 7, 8, 9,
                                         10, 11, 12, 13, 14, 15);
    }
    // B fragments: column = lane%16, K {0..7,16..23} / {8..15,24..31}.
    const int bn  = wcol + (lane & 15);
    const int bkb = (lane < 16) ? 0 : 8;
    bf16x16 bfrag[2];
#pragma unroll
    for (int j = 0; j < 2; ++j) {
      const __bf16* p = &Bs[cur][(bn + j * 16) * LDT + bkb];
      bf16x8 lo = *(const bf16x8*)(p);
      bf16x8 hi = *(const bf16x8*)(p + 16);
      bfrag[j] = __builtin_shufflevector(lo, hi, 0, 1, 2, 3, 4, 5, 6, 7, 8, 9,
                                         10, 11, 12, 13, 14, 15);
    }
#pragma unroll
    for (int i = 0; i < 4; ++i)
#pragma unroll
      for (int j = 0; j < 2; ++j)
        acc[i][j] = __builtin_amdgcn_wmma_f32_16x16x32_bf16(
            false, afrag[i], false, bfrag[j], (short)0, acc[i][j], false,
            false);
#if GEMM_ASYNC
    if (more) __builtin_amdgcn_s_wait_asynccnt(0);
#else
    if (more) store_tile(cur ^ 1);
#endif
    __syncthreads();
    cur ^= 1;
  }

  // Store per ISA C/D layout: VGPR r -> row r (lanes 0-15) / row 8+r (16-31).
  // Compile-time M => all offsets fold into immediate store offsets.
  const int crow = (lane >> 4) << 3;
  const int ccol = lane & 15;
  const size_t base = (size_t)(row0 + wrow + crow) * M + col0 + wcol + ccol;
  float*  Outf = (float*)OutP + base;
  __bf16* Outb = (__bf16*)OutP + base;
  const float* r0p = res0 ? res0 + base : nullptr;
  const float* r1p = res1 ? res1 + base : nullptr;
#pragma unroll
  for (int i = 0; i < 4; ++i) {
#pragma unroll
    for (int j = 0; j < 2; ++j) {
#pragma unroll
      for (int rr = 0; rr < 8; ++rr) {
        const float v = acc[i][j][rr];
        const size_t off = (size_t)(i * 16 + rr) * M + j * 16;
        if constexpr (EPI == 0) Outf[off] = v;
        else if constexpr (EPI == 1) Outf[off] = v / (1.f + expf(-v));
        else if constexpr (EPI == 2) Outf[off] = 1.f / (1.f + expf(-v));
        else if constexpr (EPI == 3) {
          const float t = v > 0.f ? v : 0.f;
          Outb[off] = (__bf16)(t * t);
        } else if constexpr (EPI == 4) Outf[off] = r0p[off] + v;
        else Outf[off] = r0p[off] + r1p[off] * v;
      }
    }
  }
}

// ---------------- weight transpose + f32 -> bf16 ---------------------------
__global__ void wtransp_kernel(const float* __restrict__ W,
                               __bf16* __restrict__ Wt, int K, int M) {
  const int i = blockIdx.x * 256 + threadIdx.x;
  if (i >= K * M) return;
  const int m = i / K, k = i - m * K;
  Wt[i] = (__bf16)W[(size_t)k * M + m];
}

// ---------------- LayerNorm over C=1024, one block per token ---------------
__global__ __launch_bounds__(256) void ln_kernel(const float* __restrict__ x,
                                                 const float* __restrict__ s,
                                                 const float* __restrict__ b,
                                                 float* __restrict__ out) {
  __shared__ float r1[256], r2[256];
  const int n = blockIdx.x;
  const float* xp = x + (size_t)n * kC;
  float su = 0.f, sq = 0.f;
  for (int c = threadIdx.x; c < kC; c += 256) {
    const float v = xp[c];
    su += v; sq += v * v;
  }
  r1[threadIdx.x] = su; r2[threadIdx.x] = sq;
  __syncthreads();
  for (int off = 128; off > 0; off >>= 1) {
    if (threadIdx.x < off) {
      r1[threadIdx.x] += r1[threadIdx.x + off];
      r2[threadIdx.x] += r2[threadIdx.x + off];
    }
    __syncthreads();
  }
  const float mean = r1[0] / kC;
  const float rstd = rsqrtf(r2[0] / kC - mean * mean + kEps);
  float* op = out + (size_t)n * kC;
  for (int c = threadIdx.x; c < kC; c += 256)
    op[c] = (xp[c] - mean) * rstd * s[c] + b[c];
}

// ---------------- time-shift mixing: 4-way and 2-way -----------------------
__global__ void mix4_kernel(const float* __restrict__ xa,
                            const float* __restrict__ tr,
                            const float* __restrict__ tk,
                            const float* __restrict__ tv,
                            const float* __restrict__ tg,
                            __bf16* __restrict__ ar, __bf16* __restrict__ ak,
                            __bf16* __restrict__ av, __bf16* __restrict__ ag) {
  const int i = blockIdx.x * 256 + threadIdx.x;
  if (i >= kNTok * kC) return;
  const int c = i & (kC - 1);
  const int n = i >> 10;
  const int t = n & (kT - 1);
  const float cur = xa[i];
  const float prev = (t == 0) ? 0.f : xa[i - kC];
  const float sx = prev - cur;
  ar[i] = (__bf16)(cur + sx * tr[c]);
  ak[i] = (__bf16)(cur + sx * tk[c]);
  av[i] = (__bf16)(cur + sx * tv[c]);
  ag[i] = (__bf16)(cur + sx * tg[c]);
}

__global__ void mix2_kernel(const float* __restrict__ xa,
                            const float* __restrict__ tkc,
                            const float* __restrict__ trc,
                            __bf16* __restrict__ akc,
                            __bf16* __restrict__ arc) {
  const int i = blockIdx.x * 256 + threadIdx.x;
  if (i >= kNTok * kC) return;
  const int c = i & (kC - 1);
  const int n = i >> 10;
  const int t = n & (kT - 1);
  const float cur = xa[i];
  const float prev = (t == 0) ? 0.f : xa[i - kC];
  const float sx = prev - cur;
  akc[i] = (__bf16)(cur + sx * tkc[c]);
  arc[i] = (__bf16)(cur + sx * trc[c]);
}

// ---------------- kv = k * v (in place over k) -----------------------------
__global__ void kv_kernel(float* __restrict__ k, const float* __restrict__ v) {
  const int i = blockIdx.x * 256 + threadIdx.x;
  if (i < kNTok * kC) k[i] *= v[i];
}

// ---------------- WKV state recurrence: s = s*w + kv kv^T over T -----------
// One block per (b,h); state lives in registers (16 f32 per thread).
__global__ __launch_bounds__(256) void wkv_state_kernel(
    const float* __restrict__ kv, const float* __restrict__ state_in,
    const float* __restrict__ tdec, float* __restrict__ state_out) {
  const int bh = blockIdx.x;
  const int b = bh / kH, h = bh - b * kH;
  const float w = expf(tdec[h]);
  const int tid = threadIdx.x;
  const int i  = tid & 63;
  const int jb = (tid >> 6) * 16;
  float s[16];
  const float* sin = state_in + (size_t)bh * kS * kS;
#pragma unroll
  for (int jj = 0; jj < 16; ++jj) s[jj] = sin[i * kS + jb + jj];
  __shared__ float kvl[kS];
  for (int t = 0; t < kT; ++t) {
    if (tid < kS) kvl[tid] = kv[(((size_t)b * kT + t) * kH + h) * kS + tid];
    __syncthreads();
    const float kvi = kvl[i];
#pragma unroll
    for (int jj = 0; jj < 16; ++jj) s[jj] = s[jj] * w + kvi * kvl[jb + jj];
    __syncthreads();
  }
  float* so = state_out + (size_t)bh * kS * kS;
#pragma unroll
  for (int jj = 0; jj < 16; ++jj) so[i * kS + jb + jj] = s[jj];
}

// ---------------- y_t = (state_final @ kv_t)*u + kv_t, then *= r (in place) -
__global__ __launch_bounds__(256) void wkv_y_kernel(
    const float* __restrict__ kv, const float* __restrict__ state,
    const float* __restrict__ tfirst, float* __restrict__ r_io, int tchunk) {
  const int bh = blockIdx.x;
  const int b = bh / kH, h = bh - b * kH;
  const float u = expf(tfirst[h]);
  __shared__ float sst[kS * kS];
  __shared__ float kvl[4][kS];
  for (int idx = threadIdx.x; idx < kS * kS; idx += 256)
    sst[idx] = state[(size_t)bh * kS * kS + idx];
  const int tt = threadIdx.x >> 6;
  const int i  = threadIdx.x & 63;
  const int t0 = blockIdx.y * tchunk;
  __syncthreads();
  for (int t = t0; t < t0 + tchunk; t += 4) {
    const size_t base = (((size_t)b * kT + (t + tt)) * kH + h) * kS;
    kvl[tt][i] = kv[base + i];
    __syncthreads();
    float acc = 0.f;
#pragma unroll
    for (int j = 0; j < kS; ++j) acc += sst[i * kS + j] * kvl[tt][j];
    const float yv = acc * u + kvl[tt][i];
    r_io[base + i] = yv * r_io[base + i];
    __syncthreads();
  }
}

// ---------------- GroupNorm (H groups of S) + *g, emit bf16 ----------------
__global__ __launch_bounds__(256) void gn_kernel(
    const float* __restrict__ y, const float* __restrict__ g,
    const float* __restrict__ gns, const float* __restrict__ gnb,
    __bf16* __restrict__ out) {
  __shared__ float buf[kC];
  __shared__ float msh[kH], rsh[kH];
  const int n = blockIdx.x;
  const float* yp = y + (size_t)n * kC;
  for (int c = threadIdx.x; c < kC; c += 256) buf[c] = yp[c];
  __syncthreads();
  if (threadIdx.x < kH) {
    const int hg = threadIdx.x;
    float su = 0.f, sq = 0.f;
    for (int j = 0; j < kS; ++j) {
      const float v = buf[hg * kS + j];
      su += v; sq += v * v;
    }
    const float m = su / kS;
    msh[hg] = m;
    rsh[hg] = rsqrtf(sq / kS - m * m + kEps);
  }
  __syncthreads();
  for (int c = threadIdx.x; c < kC; c += 256) {
    const int hg = c >> 6;
    const float nv = (buf[c] - msh[hg]) * rsh[hg] * gns[c] + gnb[c];
    out[(size_t)n * kC + c] = (__bf16)(nv * g[(size_t)n * kC + c]);
  }
}
}  // namespace

// ---------------------------------------------------------------------------
extern "C" void kernel_launch(void* const* d_in, const int* in_sizes, int n_in,
                              void* d_out, int out_size, void* d_ws,
                              size_t ws_size, hipStream_t stream) {
  const float* x      = (const float*)d_in[0];
  const float* st0    = (const float*)d_in[1];
  const float* tm_k   = (const float*)d_in[2];
  const float* tm_v   = (const float*)d_in[3];
  const float* tm_r   = (const float*)d_in[4];
  const float* tm_g   = (const float*)d_in[5];
  const float* tm_kc  = (const float*)d_in[6];
  const float* tm_rc  = (const float*)d_in[7];
  const float* tdec   = (const float*)d_in[8];
  const float* tfirst = (const float*)d_in[9];
  const float* Wr  = (const float*)d_in[10];
  const float* Wk  = (const float*)d_in[11];
  const float* Wv  = (const float*)d_in[12];
  const float* Wg  = (const float*)d_in[13];
  const float* Wo  = (const float*)d_in[14];
  const float* gns = (const float*)d_in[15];
  const float* gnb = (const float*)d_in[16];
  const float* Wkc = (const float*)d_in[17];
  const float* Wvc = (const float*)d_in[18];
  const float* Wrc = (const float*)d_in[19];
  const float* ln1s = (const float*)d_in[20];
  const float* ln1b = (const float*)d_in[21];
  const float* ln2s = (const float*)d_in[22];
  const float* ln2b = (const float*)d_in[23];

  char* ws = (char*)d_ws;
  const size_t MB = 1u << 20;
  // bf16 transposed weights ([M x K])
  __bf16* wr_t  = (__bf16*)(ws + 0 * MB);
  __bf16* wk_t  = (__bf16*)(ws + 2 * MB);
  __bf16* wv_t  = (__bf16*)(ws + 4 * MB);
  __bf16* wg_t  = (__bf16*)(ws + 6 * MB);
  __bf16* wo_t  = (__bf16*)(ws + 8 * MB);
  __bf16* wrc_t = (__bf16*)(ws + 10 * MB);
  __bf16* wkc_t = (__bf16*)(ws + 12 * MB);   // 8 MB
  __bf16* wvc_t = (__bf16*)(ws + 20 * MB);   // 8 MB
  // bf16 GEMM A-operands (reused across phases)
  __bf16* a_r = (__bf16*)(ws + 28 * MB);     // later: ybf
  __bf16* a_k = (__bf16*)(ws + 44 * MB);     // later: a_kc
  __bf16* a_v = (__bf16*)(ws + 60 * MB);     // later: a_rc
  __bf16* a_g = (__bf16*)(ws + 76 * MB);
  // f32 activations
  float* rbuf  = (float*)(ws + 92 * MB);     // xa, then r, then y*r
  float* kbuf  = (float*)(ws + 124 * MB);    // k, then kv
  float* vbuf  = (float*)(ws + 156 * MB);    // v
  float* gbuf  = (float*)(ws + 188 * MB);    // g(silu), then xc
  float* x1    = (float*)(ws + 220 * MB);    // x after time mixing
  __bf16* kcbf = (__bf16*)(ws + 252 * MB);   // 64 MB, relu(kc)^2 in bf16
  float* rcbuf = (float*)(ws + 316 * MB);    // sigmoid(rc)

  float* out_x     = (float*)d_out;
  float* out_state = out_x + (size_t)kNTok * kC;

  const int NT = kNTok;
  const dim3 blk(256);

  // 1) weight transpose + bf16 cast (once per launch)
  wtransp_kernel<<<(kC * kC + 255) / 256, blk, 0, stream>>>(Wr,  wr_t,  kC, kC);
  wtransp_kernel<<<(kC * kC + 255) / 256, blk, 0, stream>>>(Wk,  wk_t,  kC, kC);
  wtransp_kernel<<<(kC * kC + 255) / 256, blk, 0, stream>>>(Wv,  wv_t,  kC, kC);
  wtransp_kernel<<<(kC * kC + 255) / 256, blk, 0, stream>>>(Wg,  wg_t,  kC, kC);
  wtransp_kernel<<<(kC * kC + 255) / 256, blk, 0, stream>>>(Wo,  wo_t,  kC, kC);
  wtransp_kernel<<<(kC * kC + 255) / 256, blk, 0, stream>>>(Wrc, wrc_t, kC, kC);
  wtransp_kernel<<<(kC * kFFN + 255) / 256, blk, 0, stream>>>(Wkc, wkc_t, kC, kFFN);
  wtransp_kernel<<<(kC * kFFN + 255) / 256, blk, 0, stream>>>(Wvc, wvc_t, kFFN, kC);

  // 2) LN1 -> xa (staged in rbuf), then 4-way time-shift mix to bf16
  ln_kernel<<<NT, blk, 0, stream>>>(x, ln1s, ln1b, rbuf);
  mix4_kernel<<<(NT * kC + 255) / 256, blk, 0, stream>>>(
      rbuf, tm_r, tm_k, tm_v, tm_g, a_r, a_k, a_v, a_g);

  // 3) r/k/v/g projections (WMMA)
  const dim3 gC(kC / BN, NT / BM);
  gemm_bf16_kernel<0, kC, kC><<<gC, blk, 0, stream>>>(a_r, wr_t, nullptr, nullptr, rbuf);
  gemm_bf16_kernel<0, kC, kC><<<gC, blk, 0, stream>>>(a_k, wk_t, nullptr, nullptr, kbuf);
  gemm_bf16_kernel<0, kC, kC><<<gC, blk, 0, stream>>>(a_v, wv_t, nullptr, nullptr, vbuf);
  gemm_bf16_kernel<1, kC, kC><<<gC, blk, 0, stream>>>(a_g, wg_t, nullptr, nullptr, gbuf);

  // 4) kv = k*v; WKV recurrence -> final state (straight into d_out);
  //    readout with final state; y *= r (in place)
  kv_kernel<<<(NT * kC + 255) / 256, blk, 0, stream>>>(kbuf, vbuf);
  wkv_state_kernel<<<kB * kH, blk, 0, stream>>>(kbuf, st0, tdec, out_state);
  wkv_y_kernel<<<dim3(kB * kH, kT / 128), blk, 0, stream>>>(kbuf, out_state,
                                                            tfirst, rbuf, 128);

  // 5) GroupNorm + *g -> bf16 ybf (reuse a_r); x1 = x + ybf @ Wo
  gn_kernel<<<NT, blk, 0, stream>>>(rbuf, gbuf, gns, gnb, a_r);
  gemm_bf16_kernel<4, kC, kC><<<gC, blk, 0, stream>>>(a_r, wo_t, x, nullptr, x1);

  // 6) channel mixing: LN2 -> xc (reuse gbuf), 2-way mix to bf16
  ln_kernel<<<NT, blk, 0, stream>>>(x1, ln2s, ln2b, gbuf);
  mix2_kernel<<<(NT * kC + 255) / 256, blk, 0, stream>>>(gbuf, tm_kc, tm_rc,
                                                         a_k, a_v);

  // 7) kc = relu(xkc @ Wkc)^2 -> bf16; rc = sigmoid(xrc @ Wrc);
  //    out_x = x1 + rc * (kc @ Wvc)   (all WMMA, fused epilogues)
  gemm_bf16_kernel<3, kC, kFFN><<<dim3(kFFN / BN, NT / BM), blk, 0, stream>>>(
      a_k, wkc_t, nullptr, nullptr, kcbf);
  gemm_bf16_kernel<2, kC, kC><<<gC, blk, 0, stream>>>(a_v, wrc_t, nullptr,
                                                      nullptr, rcbuf);
  gemm_bf16_kernel<5, kFFN, kC><<<dim3(kC / BN, NT / BM), blk, 0, stream>>>(
      kcbf, wvc_t, x1, rcbuf, out_x);
}
